// ECELoss_39264591020571
// MI455X (gfx1250) — matrix operand order repaired and checked by reading, not compiled, not run on hardware
//
#include <hip/hip_runtime.h>
#include <hip/hip_bf16.h>

typedef __attribute__((ext_vector_type(16))) _Float16 v16h;
typedef __attribute__((ext_vector_type(8)))  float    v8f;

#define NBINS 15
#define NCLS  19
#define IMG_HW (512 * 1024)   // H*W, power of two -> div/mod become shifts

// ---------------------------------------------------------------- zero ws ---
__global__ void __launch_bounds__(64) ece_zero_ws(double* ws) {
  const int t = threadIdx.x;
  if (t < 3 * NBINS) ws[t] = 0.0;
}

// ------------------------------------------------ main binning via WMMA -----
// Staged word per pixel:  [15:0] conf f16 | bit(16+bin) one-hot | [31] corr.
// One v_wmma_f32_16x16x32_f16 per wave-iteration computes, for 32 pixels:
//   D[m][n] += sum_{k : bin_k == n} {conf, corr, 1}[m]
// A = data rows (row0=conf, row1=corr, row2=const 1), B = one-hot columns.
// Invalid pixels stage 0 -> all-zero B column -> contribute nothing.
__global__ void __launch_bounds__(256) ece_bin_wmma(
    const float* __restrict__ logits, const int* __restrict__ labels,
    double* __restrict__ ws, int P) {
  __shared__ unsigned stage[8][32];      // one 32-word slab per wave32
  __shared__ float    acc[3 * NBINS];    // per-block partials

  const int tid = threadIdx.x;
  if (tid < 3 * NBINS) acc[tid] = 0.0f;
  __syncthreads();

  const unsigned lane = (unsigned)tid & 31u;
  const unsigned wv   = (unsigned)tid >> 5;
  const unsigned half = lane >> 4;       // which 16-lane half
  const unsigned sub  = lane & 15u;      // A row index / B column index

  // Loop-invariant per-lane fragment-build masks (no VCC chains in the loop).
  const unsigned maskConf = (sub == 0u) ? 0xFFFFu : 0u;   // row 0: conf bits
  const unsigned maskCorr = (sub == 1u) ? 0x3C00u : 0u;   // row 1: corr -> 1.0h
  const unsigned oneRow2  = (sub == 2u) ? 0x3C00u : 0u;   // row 2: const 1.0h
  const unsigned hotBit   = (sub < (unsigned)NBINS) ? (1u << (16u + sub)) : 0u;

  v8f cacc = {};                         // 16x16 f32 accumulator fragment

  const int stride = (int)(gridDim.x * blockDim.x);
  const int iters  = (P + stride - 1) / stride;   // uniform trip count
  int pixel = (int)(blockIdx.x * blockDim.x) + tid;

  for (int it = 0; it < iters; ++it, pixel += stride) {
    unsigned pk = 0u;                    // invalid pixel -> zero contribution
    if (pixel < P) {
      const unsigned p   = (unsigned)pixel;
      const unsigned img = p / (unsigned)IMG_HW;
      const unsigned hw  = p - img * (unsigned)IMG_HW;
      const float* ptr = logits + (size_t)img * (size_t)(NCLS * IMG_HW) + hw;
      __builtin_prefetch(ptr + stride, 0, 0);      // global_prefetch_b8
      const int lab = labels[pixel];

      // Pass 0: pull all 19 class logits into registers (one load clause).
      float v[NCLS];
#pragma unroll
      for (int cc = 0; cc < NCLS; ++cc) v[cc] = ptr[(size_t)cc * IMG_HW];

      // Pass 1: max tree (no argmax tracking needed).
      float m = fmaxf(v[0], v[1]);
#pragma unroll
      for (int cc = 2; cc < NCLS; ++cc) m = fmaxf(m, v[cc]);

      // Pass 2: sum of exps (sub + v_exp + add per class).
      float s = 0.0f;
#pragma unroll
      for (int cc = 0; cc < NCLS; ++cc) s += __expf(v[cc] - m);

      // prediction == label  <=>  logits[label] == max (ties: measure zero)
      const float vlab = ptr[(size_t)lab * IMG_HW];  // cache-hit re-read
      const unsigned corr = (vlab == m) ? 1u : 0u;

      const float conf = 1.0f / s;                   // = exp(max - lse)
      int bin = (int)ceilf(conf * (float)NBINS) - 1; // (lo, hi] bins
      bin = bin < 0 ? 0 : (bin > NBINS - 1 ? NBINS - 1 : bin);

      const _Float16 ch = (_Float16)conf;
      pk = (unsigned)__builtin_bit_cast(unsigned short, ch)
         | (1u << (16u + (unsigned)bin))
         | (corr << 31);
    }

    stage[wv][lane] = pk;
    __builtin_amdgcn_wave_barrier();     // per-wave LDS is in-order (DScnt)
    const unsigned* st = stage[wv];

    v16h a, b;
#pragma unroll
    for (int j = 0; j < 16; ++j) {
      // A (16x32 f16): lane holds row `sub`; slot j <-> K = 8*half + j (+8 if j>=8)
      const unsigned kA = (j < 8) ? (8u * half + (unsigned)j)
                                  : (8u * half + (unsigned)j + 8u);
      const unsigned wA = st[kA];        // LDS broadcast read
      const unsigned bits = (wA & maskConf)
                          | ((unsigned)((int)wA >> 31) & maskCorr)  // corr -> 0x3C00
                          | oneRow2;
      a[j] = __builtin_bit_cast(_Float16, (unsigned short)bits);

      // B (32x16 f16): lane holds column `sub`; slot j <-> K = 8*half + j (+8 if j>=8)
      const unsigned wB = st[(j < 8) ? (8u * half + (unsigned)j) : (8u * half + (unsigned)j + 8u)];
      b[j] = (wB & hotBit) ? (_Float16)1.0f : (_Float16)0.0f;
    }

    // D = A x B + C : rows {conf, corr, count} x columns {bin 0..14}
    cacc = __builtin_amdgcn_wmma_f32_16x16x32_f16(
        false, a, false, b, (short)0, cacc, false, false);
  }

  // D layout: lanes 0-15 hold column n=lane; VGPR v = row M=v.
  // Rows 0/1/2 = conf_sum / corr_sum / count for bin n.
  if (lane < (unsigned)NBINS) {
    atomicAdd(&acc[3u * lane + 0u], cacc[0]);
    atomicAdd(&acc[3u * lane + 1u], cacc[1]);
    atomicAdd(&acc[3u * lane + 2u], cacc[2]);
  }
  __syncthreads();
  if (tid < 3 * NBINS) atomicAdd(&ws[tid], (double)acc[tid]);  // f64: counts exact
}

// ---------------------------------------------------------------- finalize --
__global__ void __launch_bounds__(32) ece_final(const double* __restrict__ ws,
                                                float* __restrict__ out) {
  if (threadIdx.x == 0) {
    double total = 0.0;
    for (int b = 0; b < NBINS; ++b) total += ws[3 * b + 2];
    double ece = 0.0;
    for (int b = 0; b < NBINS; ++b) {
      const double cnt = ws[3 * b + 2];
      if (cnt > 0.0) {
        const double avg_conf = ws[3 * b + 0] / cnt;
        const double acc_b    = ws[3 * b + 1] / cnt;
        const double gap = avg_conf - acc_b;
        ece += (gap < 0.0 ? -gap : gap) * (cnt / total);
      }
    }
    out[0] = (float)ece;
  }
}

// ----------------------------------------------------------------- launch ---
extern "C" void kernel_launch(void* const* d_in, const int* in_sizes, int n_in,
                              void* d_out, int out_size, void* d_ws, size_t ws_size,
                              hipStream_t stream) {
  const float* logits = (const float*)d_in[0];
  const int*   labels = (const int*)d_in[1];
  const int P = in_sizes[1];             // 8*512*1024 pixels
  double* ws = (double*)d_ws;            // 45 doubles
  float* out = (float*)d_out;

  ece_zero_ws<<<dim3(1), dim3(64), 0, stream>>>(ws);
  ece_bin_wmma<<<dim3(512), dim3(256), 0, stream>>>(logits, labels, ws, P);
  ece_final<<<dim3(1), dim3(32), 0, stream>>>(ws, out);
}
